// SSPModel_32229434589358
// MI455X (gfx1250) — matrix operand order repaired and checked by reading, not compiled, not run on hardware
//
#include <hip/hip_runtime.h>

// ---------------- CDNA5 WMMA vector types ----------------
typedef __attribute__((ext_vector_type(16))) __bf16 v16bf;
typedef __attribute__((ext_vector_type(8)))  float  v8f;

#define IN_F   512
#define HID_F  128
#define OUT_F  40
#define NT1    8      // 128/16 column tiles (layer 1)
#define NT2    3      // 48/16 column tiles (layer 2, padded 40->48)
#define KB1    16     // 512/32 k-blocks
#define KB2    4      // 128/32 k-blocks

// ================= degree / symmetric normalization =================
__global__ void k_deg_init(float* deg, int n) {
  int i = blockIdx.x * blockDim.x + threadIdx.x;
  if (i < n) deg[i] = 1.0f;                       // self-loop contributes 1
}
__global__ void k_deg_count(const int* __restrict__ dst, float* deg, long long e) {
  long long t = (long long)blockIdx.x * blockDim.x + threadIdx.x;
  if (t < e) atomicAdd(&deg[dst[t]], 1.0f);
}
__global__ void k_dinv(float* deg, int n) {
  int i = blockIdx.x * blockDim.x + threadIdx.x;
  if (i < n) deg[i] = rsqrtf(deg[i]);             // deg >= 1 always
}

// ================= weight packing: fp32 -> bf16 in per-lane B layout =================
// B 32x16 bf16 tile layout (wave32): lane holds col n = lane&15;
// element i of the 16-half chunk is k = (lane>>4)*8 + i for i<8, 16 + (lane>>4)*8 + (i-8) for i>=8.
__global__ void k_pack_w1(const float* __restrict__ W, unsigned short* __restrict__ P) {
  int t = blockIdx.x * blockDim.x + threadIdx.x;  // ((kb32*NT1 + nt)*32 + lane)
  if (t >= KB1 * NT1 * 32) return;
  int lane = t & 31;
  int nt   = (t >> 5) % NT1;
  int kb32 = (t >> 5) / NT1;
  int r = lane & 15, half = lane >> 4;
  int n = nt * 16 + r;
  unsigned short* o = P + (size_t)t * 16;
  for (int i = 0; i < 16; ++i) {
    int k = kb32 * 32 + (i < 8 ? half * 8 + i : 16 + half * 8 + (i - 8));
    __bf16 v = (__bf16)W[(size_t)k * HID_F + n];
    o[i] = __builtin_bit_cast(unsigned short, v);
  }
}
__global__ void k_pack_w2(const float* __restrict__ W, unsigned short* __restrict__ P) {
  int t = blockIdx.x * blockDim.x + threadIdx.x;
  if (t >= KB2 * NT2 * 32) return;
  int lane = t & 31;
  int nt   = (t >> 5) % NT2;
  int kb32 = (t >> 5) / NT2;
  int r = lane & 15, half = lane >> 4;
  int n = nt * 16 + r;
  unsigned short* o = P + (size_t)t * 16;
  for (int i = 0; i < 16; ++i) {
    int k = kb32 * 32 + (i < 8 ? half * 8 + i : 16 + half * 8 + (i - 8));
    float f = (n < OUT_F) ? W[(size_t)k * OUT_F + n] : 0.0f;
    __bf16 v = (__bf16)f;
    o[i] = __builtin_bit_cast(unsigned short, v);
  }
}

// Build A 16x32 bf16 tile per-lane vector from 4x float4 fp32 loads.
__device__ __forceinline__ v16bf make_a(float4 a0, float4 a1, float4 a2, float4 a3) {
  v16bf A;
  A[0]=(__bf16)a0.x;  A[1]=(__bf16)a0.y;  A[2]=(__bf16)a0.z;  A[3]=(__bf16)a0.w;
  A[4]=(__bf16)a1.x;  A[5]=(__bf16)a1.y;  A[6]=(__bf16)a1.z;  A[7]=(__bf16)a1.w;
  A[8]=(__bf16)a2.x;  A[9]=(__bf16)a2.y;  A[10]=(__bf16)a2.z; A[11]=(__bf16)a2.w;
  A[12]=(__bf16)a3.x; A[13]=(__bf16)a3.y; A[14]=(__bf16)a3.z; A[15]=(__bf16)a3.w;
  return A;
}

// ================= GEMM1: H1[N,128] = X[N,512] @ W1  (one wave = 16 rows x 128 cols) =================
__global__ void k_gemm1(const float* __restrict__ X, const unsigned short* __restrict__ W1p,
                        float* __restrict__ H1, int N) {
  int wave = (blockIdx.x * blockDim.x + threadIdx.x) >> 5;
  int lane = threadIdx.x & 31;
  int row0 = wave * 16;
  if (row0 >= N) return;
  int r = lane & 15, half = lane >> 4;
  int arow = min(row0 + r, N - 1);
  const float* a = X + (size_t)arow * IN_F;
  v8f acc[NT1] = {};
  for (int kb = 0; kb < IN_F; kb += 32) {
    const float* ap = a + kb + half * 8;
    float4 a0 = *(const float4*)(ap + 0);
    float4 a1 = *(const float4*)(ap + 4);
    float4 a2 = *(const float4*)(ap + 16);
    float4 a3 = *(const float4*)(ap + 20);
    if (kb + 32 < IN_F) __builtin_prefetch(ap + 32, 0, 0);   // global_prefetch_b8
    v16bf A = make_a(a0, a1, a2, a3);
    int kb32 = kb >> 5;
#pragma unroll
    for (int nt = 0; nt < NT1; ++nt) {
      v16bf B = *(const v16bf*)(W1p + (size_t)((kb32 * NT1 + nt) * 32 + lane) * 16);
      acc[nt] = __builtin_amdgcn_wmma_f32_16x16x32_bf16(
          false, A, false, B, (short)0, acc[nt], false, false);
    }
  }
#pragma unroll
  for (int nt = 0; nt < NT1; ++nt)
#pragma unroll
    for (int j = 0; j < 8; ++j) {
      int row = row0 + half * 8 + j;
      if (row < N) H1[(size_t)row * HID_F + nt * 16 + r] = acc[nt][j];
    }
}

// ================= GEMM2: H2[N,40] = H[N,128] @ W2  (16 rows x 48 cols padded) =================
__global__ void k_gemm2(const float* __restrict__ H, const unsigned short* __restrict__ W2p,
                        float* __restrict__ H2, int N) {
  int wave = (blockIdx.x * blockDim.x + threadIdx.x) >> 5;
  int lane = threadIdx.x & 31;
  int row0 = wave * 16;
  if (row0 >= N) return;
  int r = lane & 15, half = lane >> 4;
  int arow = min(row0 + r, N - 1);
  const float* a = H + (size_t)arow * HID_F;
  v8f acc[NT2] = {};
  for (int kb = 0; kb < HID_F; kb += 32) {
    const float* ap = a + kb + half * 8;
    float4 a0 = *(const float4*)(ap + 0);
    float4 a1 = *(const float4*)(ap + 4);
    float4 a2 = *(const float4*)(ap + 16);
    float4 a3 = *(const float4*)(ap + 20);
    v16bf A = make_a(a0, a1, a2, a3);
    int kb32 = kb >> 5;
#pragma unroll
    for (int nt = 0; nt < NT2; ++nt) {
      v16bf B = *(const v16bf*)(W2p + (size_t)((kb32 * NT2 + nt) * 32 + lane) * 16);
      acc[nt] = __builtin_amdgcn_wmma_f32_16x16x32_bf16(
          false, A, false, B, (short)0, acc[nt], false, false);
    }
  }
#pragma unroll
  for (int nt = 0; nt < NT2; ++nt)
#pragma unroll
    for (int j = 0; j < 8; ++j) {
      int row = row0 + half * 8 + j;
      int n = nt * 16 + r;
      if (row < N && n < OUT_F) H2[(size_t)row * OUT_F + n] = acc[nt][j];
    }
}

// ================= propagation: self-loop init, edge scatter =================
__global__ void k_self1(const float* __restrict__ h, const float* __restrict__ dinv,
                        float* __restrict__ agg, int n) {
  long long g = (long long)blockIdx.x * blockDim.x + threadIdx.x;  // N*32 float4 groups
  if (g >= (long long)n * 32) return;
  int i = (int)(g >> 5);
  int c = (int)(g & 31) * 4;
  float s = dinv[i] * dinv[i];
  float4 v = *(const float4*)(h + (size_t)i * HID_F + c);
  float4 o = make_float4(v.x * s, v.y * s, v.z * s, v.w * s);
  *(float4*)(agg + (size_t)i * HID_F + c) = o;
}
__global__ void k_scatter1(const int* __restrict__ src, const int* __restrict__ dst,
                           const float* __restrict__ dinv, const float* __restrict__ h,
                           float* __restrict__ agg, long long e) {
  long long t = (long long)blockIdx.x * blockDim.x + threadIdx.x;  // 32 lanes/edge, float4 each
  long long ed = t >> 5;
  if (ed >= e) return;
  int c = (int)(t & 31) * 4;
  int s = src[ed], d = dst[ed];
  float nrm = dinv[s] * dinv[d];
  float4 v = *(const float4*)(h + (size_t)s * HID_F + c);
  float* o = agg + (size_t)d * HID_F + c;
  atomicAdd(o + 0, v.x * nrm);
  atomicAdd(o + 1, v.y * nrm);
  atomicAdd(o + 2, v.z * nrm);
  atomicAdd(o + 3, v.w * nrm);
}
__global__ void k_bias_relu(float* __restrict__ agg, const float* __restrict__ b, int n) {
  long long g = (long long)blockIdx.x * blockDim.x + threadIdx.x;
  if (g >= (long long)n * 32) return;
  int i = (int)(g >> 5);
  int c = (int)(g & 31) * 4;
  float4 v  = *(const float4*)(agg + (size_t)i * HID_F + c);
  float4 bb = *(const float4*)(b + c);
  v.x = fmaxf(v.x + bb.x, 0.f); v.y = fmaxf(v.y + bb.y, 0.f);
  v.z = fmaxf(v.z + bb.z, 0.f); v.w = fmaxf(v.w + bb.w, 0.f);
  *(float4*)(agg + (size_t)i * HID_F + c) = v;
}
__global__ void k_self2(const float* __restrict__ h, const float* __restrict__ dinv,
                        float* __restrict__ agg, int n) {
  long long g = (long long)blockIdx.x * blockDim.x + threadIdx.x;  // N*10 float4 groups
  if (g >= (long long)n * 10) return;
  int i = (int)(g / 10);
  int c = (int)(g % 10) * 4;
  float s = dinv[i] * dinv[i];
  float4 v = *(const float4*)(h + (size_t)i * OUT_F + c);
  float4 o = make_float4(v.x * s, v.y * s, v.z * s, v.w * s);
  *(float4*)(agg + (size_t)i * OUT_F + c) = o;
}
__global__ void k_scatter2(const int* __restrict__ src, const int* __restrict__ dst,
                           const float* __restrict__ dinv, const float* __restrict__ h,
                           float* __restrict__ agg, long long e) {
  long long t = (long long)blockIdx.x * blockDim.x + threadIdx.x;  // 10 lanes/edge
  long long ed = t / 10;
  if (ed >= e) return;
  int c = (int)(t % 10) * 4;
  int s = src[ed], d = dst[ed];
  float nrm = dinv[s] * dinv[d];
  float4 v = *(const float4*)(h + (size_t)s * OUT_F + c);
  float* o = agg + (size_t)d * OUT_F + c;
  atomicAdd(o + 0, v.x * nrm);
  atomicAdd(o + 1, v.y * nrm);
  atomicAdd(o + 2, v.z * nrm);
  atomicAdd(o + 3, v.w * nrm);
}

// ================= bias + log_softmax (one wave32 per node, 40 classes) =================
__global__ void k_logsoftmax(const float* __restrict__ agg, const float* __restrict__ b,
                             float* __restrict__ out, int n) {
  int wave = (blockIdx.x * blockDim.x + threadIdx.x) >> 5;
  int lane = threadIdx.x & 31;
  if (wave >= n) return;
  const float* row = agg + (size_t)wave * OUT_F;
  bool has2 = lane < (OUT_F - 32);
  float a0 = row[lane] + b[lane];
  float a1 = has2 ? (row[lane + 32] + b[lane + 32]) : -INFINITY;
  float m = fmaxf(a0, a1);
  for (int off = 16; off > 0; off >>= 1) m = fmaxf(m, __shfl_xor(m, off, 32));
  float s = __expf(a0 - m) + (has2 ? __expf(a1 - m) : 0.f);
  for (int off = 16; off > 0; off >>= 1) s += __shfl_xor(s, off, 32);
  float lse = m + __logf(s);
  out[(size_t)wave * OUT_F + lane] = a0 - lse;
  if (has2) out[(size_t)wave * OUT_F + lane + 32] = a1 - lse;
}

// ================= host-side orchestration =================
static inline long long cdiv(long long a, long long b) { return (a + b - 1) / b; }

extern "C" void kernel_launch(void* const* d_in, const int* in_sizes, int n_in,
                              void* d_out, int out_size, void* d_ws, size_t ws_size,
                              hipStream_t stream) {
  const float* x  = (const float*)d_in[0];
  const int*   ei = (const int*)d_in[1];   // edge_index [2, E] (int32)
  const float* W1 = (const float*)d_in[2];
  const float* b1 = (const float*)d_in[3];
  const float* W2 = (const float*)d_in[4];
  const float* b2 = (const float*)d_in[5];
  float* out = (float*)d_out;

  const int N = in_sizes[0] / IN_F;
  const long long E = (long long)in_sizes[1] / 2;
  const int* srcI = ei;
  const int* dstI = ei + E;

  // workspace carve-out (256B aligned slices)
  char* ws = (char*)d_ws;
  size_t off = 0;
  auto carve = [&](size_t bytes) {
    void* p = ws + off;
    off = (off + bytes + 255) & ~(size_t)255;
    return p;
  };
  float* dinv = (float*)carve((size_t)N * 4);
  float* h1   = (float*)carve((size_t)N * HID_F * 4);
  float* agg1 = (float*)carve((size_t)N * HID_F * 4);
  float* h2   = (float*)carve((size_t)N * OUT_F * 4);
  float* agg2 = (float*)carve((size_t)N * OUT_F * 4);
  unsigned short* W1p = (unsigned short*)carve((size_t)KB1 * NT1 * 32 * 16 * 2);
  unsigned short* W2p = (unsigned short*)carve((size_t)KB2 * NT2 * 32 * 16 * 2);
  (void)ws_size; (void)n_in; (void)out_size;

  const int B = 256;
  long long waves = cdiv(N, 16);

  // normalization
  k_deg_init <<<cdiv(N, B), B, 0, stream>>>(dinv, N);
  k_deg_count<<<cdiv(E, B), B, 0, stream>>>(dstI, dinv, E);
  k_dinv     <<<cdiv(N, B), B, 0, stream>>>(dinv, N);

  // weight packing (bf16, WMMA B layout)
  k_pack_w1<<<cdiv(KB1 * NT1 * 32, B), B, 0, stream>>>(W1, W1p);
  k_pack_w2<<<cdiv(KB2 * NT2 * 32, B), B, 0, stream>>>(W2, W2p);

  // layer 1: GEMM -> self -> scatter -> bias+relu (in place in agg1)
  k_gemm1    <<<cdiv(waves * 32, B), B, 0, stream>>>(x, W1p, h1, N);
  k_self1    <<<cdiv((long long)N * 32, B), B, 0, stream>>>(h1, dinv, agg1, N);
  k_scatter1 <<<cdiv(E * 32, B), B, 0, stream>>>(srcI, dstI, dinv, h1, agg1, E);
  k_bias_relu<<<cdiv((long long)N * 32, B), B, 0, stream>>>(agg1, b1, N);

  // layer 2: GEMM -> self -> scatter -> bias + log_softmax
  k_gemm2    <<<cdiv(waves * 32, B), B, 0, stream>>>(agg1, W2p, h2, N);
  k_self2    <<<cdiv((long long)N * 10, B), B, 0, stream>>>(h2, dinv, agg2, N);
  k_scatter2 <<<cdiv(E * 10, B), B, 0, stream>>>(srcI, dstI, dinv, h2, agg2, E);
  k_logsoftmax<<<cdiv((long long)N * 32, B), B, 0, stream>>>(agg2, b2, out, N);
}
